// NNFM_Loss_MemoryEfficient_51573967290835
// MI455X (gfx1250) — matrix operand order repaired and checked by reading, not compiled, not run on hardware
//
#include <hip/hip_runtime.h>
#include <hip/hip_bf16.h>

typedef _Float16 v16h __attribute__((ext_vector_type(16)));
typedef float    v8f  __attribute__((ext_vector_type(8)));

#define N_PIX 16384   // 128*128
#define C_DIM 768

// ---------------------------------------------------------------------------
// Normalize one tensor along C and emit pixel-major f16 [N_PIX][C_DIM].
// permuteA!=0: swap K-bits 3<->4 inside each 32-chunk so a WMMA A-fragment is
// one contiguous 32B per-lane load (same addressing as the B fragment).
// ---------------------------------------------------------------------------
__global__ void nnfm_normalize(const float* __restrict__ F,
                               _Float16* __restrict__ out,
                               int permuteA)
{
    __shared__ float    ssum[8][32];
    __shared__ float    sinv[32];
    __shared__ _Float16 tile[32][33];

    const int tx = threadIdx.x;           // pixel within tile (0..31)
    const int ty = threadIdx.y;           // 0..7
    const int n0 = blockIdx.x * 32;
    const int n  = n0 + tx;

    // Pass 1: sum of squares over C, coalesced across pixels.
    float s = 0.0f;
    for (int c = ty; c < C_DIM; c += 8) {
        float v = F[(size_t)c * N_PIX + n];
        s += v * v;
    }
    ssum[ty][tx] = s;
    __syncthreads();
    if (ty == 0) {
        float t = 0.0f;
#pragma unroll
        for (int k = 0; k < 8; ++k) t += ssum[k][tx];
        sinv[tx] = 1.0f / fmaxf(sqrtf(t), 1e-12f);
    }
    __syncthreads();

    // storage position for original K index (c0 + tx): swap bits 3 and 4
    const int txp = permuteA ? ((tx & 7) | ((tx & 8) << 1) | ((tx & 16) >> 1))
                             : tx;

    // Pass 2: scale, transpose 32x32 tiles through LDS, write [n][c] f16.
    for (int c0 = 0; c0 < C_DIM; c0 += 32) {
        for (int cy = ty; cy < 32; cy += 8)
            tile[cy][tx] = (_Float16)(F[(size_t)(c0 + cy) * N_PIX + n] * sinv[tx]);
        __syncthreads();
        for (int ry = ty; ry < 32; ry += 8)
            out[(size_t)(n0 + ry) * C_DIM + c0 + txp] = tile[tx][ry];
        __syncthreads();
    }
}

// ---------------------------------------------------------------------------
// Init row maxima to the minimal monotonic encoding.
// ---------------------------------------------------------------------------
__global__ void nnfm_init(unsigned int* __restrict__ rowmax)
{
    int n = blockIdx.x * 256 + threadIdx.x;
    if (n < N_PIX) rowmax[n] = 0u;
}

// ---------------------------------------------------------------------------
// One wave per 64x64 tile of sim = A * B^T (K = 768), f16 WMMA, f32 accum.
// 4x4 fragment accumulators; row-wise max folded into rowmax via atomicMax
// on a monotonic float->uint encoding.
// ---------------------------------------------------------------------------
__global__ __launch_bounds__(32)
void nnfm_gemm_max(const _Float16* __restrict__ A,
                   const _Float16* __restrict__ B,
                   unsigned int* __restrict__ rowmax)
{
    const int lane = threadIdx.x & 31;
    const int r0   = blockIdx.y * 64;
    const int s0   = blockIdx.x * 64;
    const int mRow = lane & 15;       // row (A) / column (B) within fragment
    const int hi   = lane >> 4;       // lane-group: selects K half
    const size_t kOff = (size_t)(hi * 16);

    const _Float16* aPtr[4];
    const _Float16* bPtr[4];
#pragma unroll
    for (int i = 0; i < 4; ++i)
        aPtr[i] = A + (size_t)(r0 + 16 * i + mRow) * C_DIM + kOff;
#pragma unroll
    for (int j = 0; j < 4; ++j)
        bPtr[j] = B + (size_t)(s0 + 16 * j + mRow) * C_DIM + kOff;

    const v8f vzero = {0.f, 0.f, 0.f, 0.f, 0.f, 0.f, 0.f, 0.f};
    v8f acc[4][4];
#pragma unroll
    for (int i = 0; i < 4; ++i)
#pragma unroll
        for (int j = 0; j < 4; ++j) acc[i][j] = vzero;

    for (int k0 = 0; k0 < C_DIM; k0 += 32) {
        v16h a[4], b[4];
#pragma unroll
        for (int i = 0; i < 4; ++i) a[i] = *(const v16h*)(aPtr[i] + k0);
#pragma unroll
        for (int j = 0; j < 4; ++j) b[j] = *(const v16h*)(bPtr[j] + k0);
#pragma unroll
        for (int i = 0; i < 4; ++i)
#pragma unroll
            for (int j = 0; j < 4; ++j)
                acc[i][j] = __builtin_amdgcn_wmma_f32_16x16x32_f16(
                    /*neg_a=*/false, a[i], /*neg_b=*/false, b[j],
                    /*c_mod=*/(short)0, acc[i][j],
                    /*reuse_a=*/false, /*reuse_b=*/false);
    }

    // Row-wise max: fold s-fragments, then reduce across the 16-lane half.
    // D layout: lane holds column N = lane&15; VGPR v holds row M = v + 8*hi.
#pragma unroll
    for (int i = 0; i < 4; ++i) {
        v8f m = acc[i][0];
#pragma unroll
        for (int j = 1; j < 4; ++j)
#pragma unroll
            for (int v = 0; v < 8; ++v) m[v] = fmaxf(m[v], acc[i][j][v]);
#pragma unroll
        for (int v = 0; v < 8; ++v) {
            float x = m[v];
            x = fmaxf(x, __shfl_xor(x, 1, 32));
            x = fmaxf(x, __shfl_xor(x, 2, 32));
            x = fmaxf(x, __shfl_xor(x, 4, 32));
            x = fmaxf(x, __shfl_xor(x, 8, 32));
            if (mRow == 0) {
                unsigned bits = __float_as_uint(x);
                unsigned enc  = (bits & 0x80000000u) ? ~bits
                                                     : (bits | 0x80000000u);
                atomicMax(&rowmax[r0 + 16 * i + 8 * hi + v], enc);
            }
        }
    }
}

// ---------------------------------------------------------------------------
// loss = mean(1 - rowmax)
// ---------------------------------------------------------------------------
__global__ void nnfm_finalize(const unsigned int* __restrict__ rowmax,
                              float* __restrict__ out)
{
    __shared__ float red[256];
    float s = 0.0f;
    for (int n = threadIdx.x; n < N_PIX; n += 256) {
        unsigned enc  = rowmax[n];
        unsigned bits = (enc & 0x80000000u) ? (enc & 0x7fffffffu) : ~enc;
        s += 1.0f - __uint_as_float(bits);
    }
    red[threadIdx.x] = s;
    __syncthreads();
    for (int off = 128; off > 0; off >>= 1) {
        if (threadIdx.x < off) red[threadIdx.x] += red[threadIdx.x + off];
        __syncthreads();
    }
    if (threadIdx.x == 0) out[0] = red[0] / (float)N_PIX;
}

// ---------------------------------------------------------------------------
extern "C" void kernel_launch(void* const* d_in, const int* in_sizes, int n_in,
                              void* d_out, int out_size, void* d_ws, size_t ws_size,
                              hipStream_t stream)
{
    (void)in_sizes; (void)n_in; (void)out_size; (void)ws_size;

    const float* F_r = (const float*)d_in[0];
    const float* F_s = (const float*)d_in[1];
    float*       out = (float*)d_out;

    _Float16* A = (_Float16*)d_ws;                       // 16384*768 f16 = 24 MB
    _Float16* B = A + (size_t)N_PIX * C_DIM;             // 24 MB
    unsigned int* rowmax = (unsigned int*)(B + (size_t)N_PIX * C_DIM);  // 64 KB

    nnfm_init<<<N_PIX / 256, 256, 0, stream>>>(rowmax);
    nnfm_normalize<<<N_PIX / 32, dim3(32, 8), 0, stream>>>(F_r, A, /*permuteA=*/0);
    nnfm_normalize<<<N_PIX / 32, dim3(32, 8), 0, stream>>>(F_s, B, /*permuteA=*/0);
    nnfm_gemm_max<<<dim3(N_PIX / 64, N_PIX / 64), 32, 0, stream>>>(A, B, rowmax);
    nnfm_finalize<<<1, 256, 0, stream>>>(rowmax, out);
}
